// GatedTrajectoryTransformer_67216238182700
// MI455X (gfx1250) — matrix operand order repaired and checked by reading, not compiled, not run on hardware
//
#include <hip/hip_runtime.h>
#include <math.h>

// ---------------------------------------------------------------------------
// Types for CDNA5 WMMA / TDM (gfx1250, wave32)
// ---------------------------------------------------------------------------
typedef __bf16          v16bf __attribute__((ext_vector_type(16)));
typedef unsigned short  v16u  __attribute__((ext_vector_type(16)));
typedef float           v8f   __attribute__((ext_vector_type(8)));
typedef unsigned int    v4u   __attribute__((ext_vector_type(4)));
typedef int             v8i   __attribute__((ext_vector_type(8)));
typedef int             v4i   __attribute__((ext_vector_type(4)));

union U16 { v16u u; v16bf b; };

__device__ __forceinline__ unsigned short f2bf(float f) {
    unsigned u = __float_as_uint(f);
    unsigned r = u + 0x7FFFu + ((u >> 16) & 1u);   // round-to-nearest-even
    return (unsigned short)(r >> 16);
}

__device__ __forceinline__ v8f wmma_bf16(const U16& a, const U16& b, v8f c) {
    // D = A(16x32 bf16) * B(32x16 bf16) + C(16x16 f32)
    return __builtin_amdgcn_wmma_f32_16x16x32_bf16(
        false, a.b, false, b.b, (short)0, c, false, false);
}

__device__ __forceinline__ float sigmoidf_(float x) {
    return 1.0f / (1.0f + __expf(-x));
}

// ---------------------------------------------------------------------------
// TDM: async 2-D tile load (fp32), 64 rows x 32 cols, into LDS with hardware
// padding of 1 DWORD per 32-DWORD row (-> 33-float LDS row stride, conflict
// free). Rows beyond rows_rem are zero-filled by TDM OOB handling.
// D# layout per CDNA5 ISA 8.3/8.4 (group0 128b, group1 256b; groups 2/3 zero).
// This toolchain exposes the 6-arg builtin (extra int32x8 group + cpol).
// ---------------------------------------------------------------------------
__device__ __forceinline__ void tdm_load_tile(unsigned lds_off, const float* gptr,
                                              int rows_rem, int cols_rem,
                                              int rowStrideElems)
{
    const unsigned long long ga = (unsigned long long)(size_t)gptr;
    const unsigned long long s0 = (unsigned long long)rowStrideElems;

    v4u g0;
    g0[0] = 1u;                                        // count=1, user mode
    g0[1] = lds_off;                                   // lds_addr
    g0[2] = (unsigned)(ga & 0xFFFFFFFFull);            // global_addr[31:0]
    g0[3] = (unsigned)((ga >> 32) & 0x1FFFFFFull)      // global_addr[56:32]
          | (2u << 30);                                // type = 2 ("image")

    v8i g1;
    g1[0] = (int)((2u << 16)      // data_size = 4 bytes
                | (1u << 20)      // pad_enable
                | (4u << 22));    // pad_interval: 32 DWORDs ; pad_amount: 1 DWORD
    g1[1] = (int)(((unsigned)cols_rem & 0xFFFFu) << 16);                // dim0 lo
    g1[2] = (int)((((unsigned)cols_rem >> 16) & 0xFFFFu)
                | (((unsigned)rows_rem & 0xFFFFu) << 16));              // dim0 hi | dim1 lo
    g1[3] = (int)((((unsigned)rows_rem >> 16) & 0xFFFFu)
                | (32u << 16));                                         // dim1 hi | tile_dim0=32
    g1[4] = (int)64u;                                                   // tile_dim1=64, tile_dim2=0
    g1[5] = (int)(unsigned)(s0 & 0xFFFFFFFFull);                        // dim0_stride lo
    g1[6] = (int)(unsigned)((s0 >> 32) & 0xFFFFull);                    // dim0_stride hi | dim1_stride lo
    g1[7] = 0;                                                          // dim1_stride hi

    v4i g2 = {0, 0, 0, 0};
    v4i g3 = {0, 0, 0, 0};
    v8i g4 = {0, 0, 0, 0, 0, 0, 0, 0};
    __builtin_amdgcn_tensor_load_to_lds(g0, g1, g2, g3, g4, 0);
}

// ---------------------------------------------------------------------------
// WMMA GEMM with TDM double-buffered tile staging:
//   C[M,N] = A[M,K] * W[N,K]^T (+ bias[N])
// Block: 128 threads (4 waves). Block tile 64(M) x 64(N).
// Wave w owns a 16-row strip and all 64 cols: 1 A frag x 4 B frags = 4 WMMA
// per K step. Wave 0 drives the TDM; everyone syncs at the barrier.
// Requires: N % 64 == 0, K % 32 == 0. M arbitrary (TDM OOB zero-fills rows).
// ---------------------------------------------------------------------------
__global__ __launch_bounds__(128)
void k_gemm(const float* __restrict__ A, const float* __restrict__ W,
            const float* __restrict__ bias, float* __restrict__ C,
            int M, int N, int K)
{
    __shared__ float As[2][64 * 33];
    __shared__ float Bs[2][64 * 33];

    const int tid  = threadIdx.x;
    const int w    = tid >> 5;
    const int lane = tid & 31;
    const int half = lane >> 4;
    const int l16  = lane & 15;

    const int blockM = blockIdx.y * 64;
    const int blockN = blockIdx.x * 64;

    v8f acc0 = {}, acc1 = {}, acc2 = {}, acc3 = {};

    // prologue: stage first K tile into buffer 0
    if (w == 0) {
        tdm_load_tile((unsigned)(size_t)(&As[0][0]),
                      A + (size_t)blockM * K, M - blockM, K, K);
        tdm_load_tile((unsigned)(size_t)(&Bs[0][0]),
                      W + (size_t)blockN * K, N - blockN, K, K);
    }

    for (int k0 = 0; k0 < K; k0 += 32) {
        const int buf = (k0 >> 5) & 1;
        if (w == 0) {
            if (k0 + 32 < K) {
                // issue next tile into the other buffer, then wait for current
                tdm_load_tile((unsigned)(size_t)(&As[buf ^ 1][0]),
                              A + (size_t)blockM * K + (k0 + 32),
                              M - blockM, K - (k0 + 32), K);
                tdm_load_tile((unsigned)(size_t)(&Bs[buf ^ 1][0]),
                              W + (size_t)blockN * K + (k0 + 32),
                              N - blockN, K - (k0 + 32), K);
                __builtin_amdgcn_s_wait_tensorcnt((short)2);
            } else {
                __builtin_amdgcn_s_wait_tensorcnt((short)0);
            }
        }
        __syncthreads();   // current buffer valid for all waves

        // ---- A fragment (16x32, contiguous k = 16*half + s), fp32 LDS -> bf16 ----
        U16 af;
        {
            const float* ab = &As[buf][(w * 16 + l16) * 33];
            #pragma unroll
            for (int i = 0; i < 8; ++i) {
                const int k = half * 16 + 2 * i;
                af.u[2 * i]     = f2bf(ab[k]);
                af.u[2 * i + 1] = f2bf(ab[k + 1]);
            }
        }
        // ---- 4 B fragments (32x16, contiguous k = 16*half + s, same order as A) ----
        #pragma unroll
        for (int nt = 0; nt < 4; ++nt) {
            U16 bf;
            const float* bb = &Bs[buf][(nt * 16 + l16) * 33];
            #pragma unroll
            for (int i = 0; i < 8; ++i) {
                const int k = half * 16 + 2 * i;
                bf.u[2 * i]     = f2bf(bb[k]);
                bf.u[2 * i + 1] = f2bf(bb[k + 1]);
            }
            if      (nt == 0) acc0 = wmma_bf16(af, bf, acc0);
            else if (nt == 1) acc1 = wmma_bf16(af, bf, acc1);
            else if (nt == 2) acc2 = wmma_bf16(af, bf, acc2);
            else              acc3 = wmma_bf16(af, bf, acc3);
        }
        __syncthreads();   // all reads done before wave 0 overwrites this buffer
    }

    // ---- epilogue ----
    #pragma unroll
    for (int nt = 0; nt < 4; ++nt) {
        v8f cc = (nt == 0) ? acc0 : (nt == 1) ? acc1 : (nt == 2) ? acc2 : acc3;
        const int col = blockN + nt * 16 + l16;
        const float bv = bias ? bias[col] : 0.0f;
        #pragma unroll
        for (int j = 0; j < 8; ++j) {
            const int row = blockM + w * 16 + half * 8 + j;
            if (row < M) C[(size_t)row * N + col] = cc[j] + bv;
        }
    }
}

// ---------------------------------------------------------------------------
// Fused GRU recurrent step (one launch per timestep):
//   gh = h_in @ Whh^T + bhh ; gates ; h_out = (1-z)*n + z*h_in ; y_t = h_out
// Grid: 8 blocks x 128 threads. Block owns 64 hidden columns, wave owns 16.
// h ping-pongs across launches (no intra-launch read/write race).
// B=32, E=512, gates N=1536, K=512 fixed. Whh is L2-resident (3 MB).
// ---------------------------------------------------------------------------
__global__ __launch_bounds__(128)
void k_gru_step(const float* __restrict__ hin, float* __restrict__ hout,
                const float* __restrict__ gi, long giStrideB,
                const float* __restrict__ Whh, const float* __restrict__ bhh,
                float* __restrict__ y, long yStrideB)
{
    __shared__ unsigned short hs[32 * 520];   // h as bf16, padded rows

    const int tid = threadIdx.x;
    for (int idx = tid; idx < 32 * 512; idx += 128) {
        const int m = idx >> 9, c = idx & 511;
        hs[m * 520 + c] = f2bf(hin[idx]);
    }
    __syncthreads();

    const int w    = tid >> 5;
    const int lane = tid & 31;
    const int half = lane >> 4;
    const int l16  = lane & 15;
    const int col  = blockIdx.x * 64 + w * 16 + l16;

    for (int mhalf = 0; mhalf < 2; ++mhalf) {
        v8f cr = {}, cz = {}, cn = {};
        for (int k0 = 0; k0 < 512; k0 += 32) {
            U16 af;
            const int m = mhalf * 16 + l16;
            #pragma unroll
            for (int i = 0; i < 8; ++i) {
                const int k = k0 + half * 16 + 2 * i;
                af.u[2 * i]     = hs[m * 520 + k];
                af.u[2 * i + 1] = hs[m * 520 + k + 1];
            }
            #pragma unroll
            for (int g = 0; g < 3; ++g) {
                U16 bf;
                const float* wr = Whh + (size_t)(g * 512 + col) * 512;
                #pragma unroll
                for (int i = 0; i < 8; ++i) {
                    const int k = k0 + half * 16 + 2 * i;
                    const float2 t = *reinterpret_cast<const float2*>(wr + k);
                    bf.u[2 * i]     = f2bf(t.x);
                    bf.u[2 * i + 1] = f2bf(t.y);
                }
                if      (g == 0) cr = wmma_bf16(af, bf, cr);
                else if (g == 1) cz = wmma_bf16(af, bf, cz);
                else             cn = wmma_bf16(af, bf, cn);
            }
        }
        // gates (purely per-lane elementwise across matching C positions)
        const float br = bhh[col], bz = bhh[512 + col], bn = bhh[1024 + col];
        #pragma unroll
        for (int j = 0; j < 8; ++j) {
            const int b = mhalf * 16 + half * 8 + j;
            const float* gip = gi + (size_t)b * giStrideB;
            const float gir = gip[col];
            const float giz = gip[512 + col];
            const float gin = gip[1024 + col];
            const float hold = hin[b * 512 + col];
            const float rr = sigmoidf_(gir + cr[j] + br);
            const float zz = sigmoidf_(giz + cz[j] + bz);
            const float nn = tanhf(gin + rr * (cn[j] + bn));
            const float hn = (1.0f - zz) * nn + zz * hold;
            hout[b * 512 + col] = hn;
            y[(size_t)b * yStrideB + col] = hn;
        }
    }
}

// ---------------------------------------------------------------------------
// Flash-style attention, one block per (batch, head). K/V in dynamic LDS.
// qkv rows: (b*Sl + s) * 1536, q at +0, k at +512, v at +1024, head h*64.
// ---------------------------------------------------------------------------
__global__ __launch_bounds__(128)
void k_attn(const float* __restrict__ qkv, float* __restrict__ out,
            int Sl, int causal)
{
    extern __shared__ float sm[];
    float* ks = sm;
    float* vs = sm + (size_t)Sl * 64;

    const int b  = blockIdx.x >> 3;
    const int hh = blockIdx.x & 7;
    const float* base = qkv + (size_t)b * Sl * 1536;

    for (int idx = threadIdx.x; idx < Sl * 64; idx += blockDim.x) {
        const int j = idx >> 6, d = idx & 63;
        ks[idx] = base[(size_t)j * 1536 + 512  + hh * 64 + d];
        vs[idx] = base[(size_t)j * 1536 + 1024 + hh * 64 + d];
    }
    __syncthreads();

    const int i = threadIdx.x;
    if (i < Sl) {
        float q[64], acc[64];
        const float* qp = base + (size_t)i * 1536 + hh * 64;
        #pragma unroll
        for (int d = 0; d < 64; ++d) { q[d] = qp[d]; acc[d] = 0.0f; }

        float mmax = -3.4e38f, lsum = 0.0f;
        const int jend = causal ? (i + 1) : Sl;
        for (int j = 0; j < jend; ++j) {
            float s = 0.0f;
            #pragma unroll
            for (int d = 0; d < 64; ++d) s += q[d] * ks[j * 64 + d];
            s *= 0.125f;                     // 1/sqrt(64)
            const float mnew = fmaxf(mmax, s);
            const float corr = __expf(mmax - mnew);
            const float p    = __expf(s - mnew);
            lsum = lsum * corr + p;
            #pragma unroll
            for (int d = 0; d < 64; ++d)
                acc[d] = acc[d] * corr + p * vs[j * 64 + d];
            mmax = mnew;
        }
        const float inv = 1.0f / lsum;
        float* op = out + ((size_t)b * Sl + i) * 512 + hh * 64;
        #pragma unroll
        for (int d = 0; d < 64; ++d) op[d] = acc[d] * inv;
    }
}

// ---------------------------------------------------------------------------
// Fused residual + LayerNorm over E=512: out = LN(a + b) * g + beta
// One block (128 threads) per row.
// ---------------------------------------------------------------------------
__global__ __launch_bounds__(128)
void k_add_ln(const float* __restrict__ a, const float* __restrict__ b,
              const float* __restrict__ g, const float* __restrict__ beta,
              float* __restrict__ out)
{
    __shared__ float sh[128];
    const int row = blockIdx.x;
    const int tid = threadIdx.x;
    const float* pa = a + (size_t)row * 512;
    const float* pb = b + (size_t)row * 512;

    float v[4];
    float s = 0.0f;
    #pragma unroll
    for (int q = 0; q < 4; ++q) {
        v[q] = pa[tid + q * 128] + pb[tid + q * 128];
        s += v[q];
    }
    sh[tid] = s; __syncthreads();
    for (int o = 64; o > 0; o >>= 1) { if (tid < o) sh[tid] += sh[tid + o]; __syncthreads(); }
    const float mean = sh[0] * (1.0f / 512.0f);
    __syncthreads();

    float s2 = 0.0f;
    #pragma unroll
    for (int q = 0; q < 4; ++q) { const float d = v[q] - mean; s2 += d * d; }
    sh[tid] = s2; __syncthreads();
    for (int o = 64; o > 0; o >>= 1) { if (tid < o) sh[tid] += sh[tid + o]; __syncthreads(); }
    const float inv = rsqrtf(sh[0] * (1.0f / 512.0f) + 1e-5f);

    #pragma unroll
    for (int q = 0; q < 4; ++q) {
        const int c = tid + q * 128;
        out[(size_t)row * 512 + c] = (v[q] - mean) * inv * g[c] + beta[c];
    }
}

// ---------------------------------------------------------------------------
// Utility kernels
// ---------------------------------------------------------------------------
__global__ void k_zero(float* p, int n)
{
    const int i = blockIdx.x * blockDim.x + threadIdx.x;
    if (i < n) p[i] = 0.0f;
}

__global__ void k_copy_rows(float* __restrict__ dst, long dstStride,
                            const float* __restrict__ src, long srcStride,
                            int rows, int cols)
{
    const int idx = blockIdx.x * blockDim.x + threadIdx.x;
    if (idx < rows * cols) {
        const int r = idx / cols, c = idx % cols;
        dst[(size_t)r * dstStride + c] = src[(size_t)r * srcStride + c];
    }
}

// gather cur[b, 0..Sl-1, :] (stored with seq stride T) -> packed (B*Sl, 512)
__global__ void k_pack(float* __restrict__ dst, const float* __restrict__ cur,
                       int B, int Sl, int T)
{
    const int idx = blockIdx.x * blockDim.x + threadIdx.x;
    if (idx < B * Sl * 512) {
        const int b = idx / (Sl * 512);
        const int rem = idx - b * Sl * 512;
        const int s = rem >> 9;
        const int c = rem & 511;
        dst[idx] = cur[((size_t)b * T + s) * 512 + c];
    }
}

__global__ __launch_bounds__(128)
void k_rownorm(const float* __restrict__ cb, float* __restrict__ c2)
{
    __shared__ float sh[128];
    const int row = blockIdx.x, tid = threadIdx.x;
    float s = 0.0f;
    for (int c = tid; c < 512; c += 128) {
        const float v = cb[(size_t)row * 512 + c];
        s += v * v;
    }
    sh[tid] = s; __syncthreads();
    for (int o = 64; o > 0; o >>= 1) { if (tid < o) sh[tid] += sh[tid + o]; __syncthreads(); }
    if (tid == 0) c2[row] = sh[0];
}

// per batch row m: argmin_n ( |cb_n|^2 - 2 * G[m,n] ), write q row and idx
__global__ __launch_bounds__(128)
void k_vq_argmin(const float* __restrict__ G, const float* __restrict__ c2,
                 const float* __restrict__ cb, float* __restrict__ q,
                 float* __restrict__ idx_out)
{
    __shared__ float bs[128];
    __shared__ int   bi[128];
    const int m = blockIdx.x, tid = threadIdx.x;

    float best = 3.4e38f; int bidx = 0;
    for (int n = tid; n < 512; n += 128) {
        const float d = c2[n] - 2.0f * G[(size_t)m * 512 + n];
        if (d < best) { best = d; bidx = n; }
    }
    bs[tid] = best; bi[tid] = bidx; __syncthreads();
    for (int o = 64; o > 0; o >>= 1) {
        if (tid < o) {
            if (bs[tid + o] < bs[tid] ||
                (bs[tid + o] == bs[tid] && bi[tid + o] < bi[tid])) {
                bs[tid] = bs[tid + o]; bi[tid] = bi[tid + o];
            }
        }
        __syncthreads();
    }
    const int sel = bi[0];
    for (int c = tid; c < 512; c += 128)
        q[(size_t)m * 512 + c] = cb[(size_t)sel * 512 + c];
    if (tid == 0) idx_out[m] = (float)sel;
}

// vq_loss = (1 + 0.25) * mean((q - e)^2) over 32x512
__global__ __launch_bounds__(256)
void k_vq_loss(const float* __restrict__ q, const float* __restrict__ e,
               float* __restrict__ out)
{
    __shared__ float sh[256];
    const int tid = threadIdx.x;
    float s = 0.0f;
    for (int i = tid; i < 32 * 512; i += 256) {
        const float d = q[i] - e[i];
        s += d * d;
    }
    sh[tid] = s; __syncthreads();
    for (int o = 128; o > 0; o >>= 1) { if (tid < o) sh[tid] += sh[tid + o]; __syncthreads(); }
    if (tid == 0) out[0] = sh[0] * (1.25f / 16384.0f);
}

// ---------------------------------------------------------------------------
// Host-side orchestration
// ---------------------------------------------------------------------------
namespace {

constexpr int Hh = 8, NL = 4, Bb = 32, Ss = 128, Tt = 16;

struct LayerW {
    const float *Wqkv, *bqkv, *Wo, *bo;
    const float *Wih1, *Whh1, *bih1, *bhh1;
    const float *Wih2, *Whh2, *bih2, *bhh2;
    const float *g1, *b1, *g2, *b2;
};

struct Ws {
    float *X0, *XA, *QKV, *ATT, *PR, *AA, *GI, *YY, *GG;
    float *H0, *H1, *CUR, *EOS, *GQ, *C2, *QV, *YL, *OO, *TE;
};

static LayerW get_layer(void* const* din, int base) {
    LayerW L;
    L.Wqkv = (const float*)din[base + 0];  L.bqkv = (const float*)din[base + 1];
    L.Wo   = (const float*)din[base + 2];  L.bo   = (const float*)din[base + 3];
    L.Wih1 = (const float*)din[base + 4];  L.Whh1 = (const float*)din[base + 5];
    L.bih1 = (const float*)din[base + 6];  L.bhh1 = (const float*)din[base + 7];
    L.Wih2 = (const float*)din[base + 8];  L.Whh2 = (const float*)din[base + 9];
    L.bih2 = (const float*)din[base + 10]; L.bhh2 = (const float*)din[base + 11];
    L.g1   = (const float*)din[base + 12]; L.b1   = (const float*)din[base + 13];
    L.g2   = (const float*)din[base + 14]; L.b2   = (const float*)din[base + 15];
    return L;
}

static void gemm(const float* A, const float* W, const float* bias, float* C,
                 int M, int N, int K, hipStream_t st) {
    dim3 g(N / 64, (M + 63) / 64);
    k_gemm<<<g, 128, 0, st>>>(A, W, bias, C, M, N, K);
}

static void gru_seq(const float* gi, const float* Whh, const float* bhh,
                    float* y, int Sl, Ws& w, hipStream_t st) {
    k_zero<<<(16384 + 255) / 256, 256, 0, st>>>(w.H0, 16384);
    float* hin = w.H0;
    float* hout = w.H1;
    for (int s = 0; s < Sl; ++s) {
        k_gru_step<<<8, 128, 0, st>>>(hin, hout,
                                      gi + (size_t)s * 1536, (long)Sl * 1536,
                                      Whh, bhh,
                                      y + (size_t)s * 512, (long)Sl * 512);
        float* t = hout; hout = hin; hin = t;
    }
}

static void run_layer(const LayerW& L, const float* xin, float* xout,
                      int Sl, int causal, Ws& w, hipStream_t st) {
    const int M = Bb * Sl;
    gemm(xin, L.Wqkv, L.bqkv, w.QKV, M, 1536, 512, st);
    const size_t shm = (size_t)Sl * 64 * 2 * sizeof(float);
    k_attn<<<Bb * Hh, 128, shm, st>>>(w.QKV, w.ATT, Sl, causal);
    gemm(w.ATT, L.Wo, L.bo, w.PR, M, 512, 512, st);
    k_add_ln<<<M, 128, 0, st>>>(w.PR, xin, L.g1, L.b1, w.AA);
    gemm(w.AA, L.Wih1, L.bih1, w.GI, M, 1536, 512, st);
    gru_seq(w.GI, L.Whh1, L.bhh1, w.YY, Sl, w, st);
    k_add_ln<<<M, 128, 0, st>>>(w.YY, w.AA, L.g2, L.b2, w.GG);
    gemm(w.GG, L.Wih2, L.bih2, w.GI, M, 1536, 512, st);
    gru_seq(w.GI, L.Whh2, L.bhh2, xout, Sl, w, st);
}

} // namespace

extern "C" void kernel_launch(void* const* d_in, const int* in_sizes, int n_in,
                              void* d_out, int out_size, void* d_ws, size_t ws_size,
                              hipStream_t stream)
{
    (void)in_sizes; (void)n_in; (void)out_size; (void)ws_size;

    // Inputs (insertion order): src, tgt, emb_W, emb_b, out_W, out_b, codebook,
    // then enc[0..3] x 16 tensors, then dec[0..3] x 16 tensors.
    const float* src   = (const float*)d_in[0];
    const float* emb_W = (const float*)d_in[2];
    const float* emb_b = (const float*)d_in[3];
    const float* out_W = (const float*)d_in[4];
    const float* out_b = (const float*)d_in[5];
    const float* cb    = (const float*)d_in[6];
    float* dout = (float*)d_out;

    // workspace layout (floats)
    float* f = (float*)d_ws;
    size_t off = 0;
    auto alloc = [&](size_t n) { float* p = f + off; off += n; return p; };
    Ws w;
    w.X0  = alloc((size_t)4096 * 512);
    w.XA  = alloc((size_t)4096 * 512);
    w.QKV = alloc((size_t)4096 * 1536);
    w.ATT = alloc((size_t)4096 * 512);
    w.PR  = alloc((size_t)4096 * 512);
    w.AA  = alloc((size_t)4096 * 512);
    w.GI  = alloc((size_t)4096 * 1536);
    w.YY  = alloc((size_t)4096 * 512);
    w.GG  = alloc((size_t)4096 * 512);
    w.H0  = alloc(16384);
    w.H1  = alloc(16384);
    w.CUR = alloc((size_t)Bb * Tt * 512);
    w.EOS = alloc(16384);
    w.GQ  = alloc(16384);
    w.C2  = alloc(512);
    w.QV  = alloc(16384);
    w.YL  = alloc(16384);
    w.OO  = alloc((size_t)Bb * 64);
    w.TE  = alloc(16384);

    // ---------------- Encoder ----------------
    // src_emb = src @ emb_W^T + emb_b : (B*S, 64) x (512, 64)^T
    gemm(src, emb_W, emb_b, w.X0, Bb * Ss, 512, 64, stream);

    float* xp = w.X0;
    float* xq = w.XA;
    for (int l = 0; l < NL; ++l) {
        LayerW L = get_layer(d_in, 7 + l * 16);
        run_layer(L, xp, xq, Ss, /*causal=*/0, w, stream);
        float* t = xp; xp = xq; xq = t;
    }
    // memory result in xp; eos = memory[:, -1, :]
    k_copy_rows<<<(Bb * 512 + 255) / 256, 256, 0, stream>>>(
        w.EOS, 512, xp + (size_t)(Ss - 1) * 512, (long)Ss * 512, Bb, 512);

    // ---------------- VQ ----------------
    k_rownorm<<<512, 128, 0, stream>>>(cb, w.C2);
    gemm(w.EOS, cb, nullptr, w.GQ, Bb, 512, 512, stream);          // eos . cb^T
    k_vq_argmin<<<Bb, 128, 0, stream>>>(w.GQ, w.C2, cb, w.QV,
                                        dout + 32 * Tt * 64 + 1); // idx floats
    k_vq_loss<<<1, 256, 0, stream>>>(w.QV, w.EOS, dout + 32 * Tt * 64);
    // q_st forward value == q ; decoder slot 0
    k_copy_rows<<<(Bb * 512 + 255) / 256, 256, 0, stream>>>(
        w.CUR, (long)Tt * 512, w.QV, 512, Bb, 512);

    // ---------------- Decoder (autoregressive, T steps) ----------------
    for (int t = 0; t < Tt; ++t) {
        const int Sl = t + 1;
        const int M = Bb * Sl;
        k_pack<<<(M * 512 + 255) / 256, 256, 0, stream>>>(w.X0, w.CUR, Bb, Sl, Tt);

        float* dp = w.X0;
        float* dq = w.XA;
        for (int l = 0; l < NL; ++l) {
            LayerW L = get_layer(d_in, 7 + (NL + l) * 16);
            run_layer(L, dp, dq, Sl, /*causal=*/1, w, stream);
            float* tt = dp; dp = dq; dq = tt;
        }
        // y[:, -1, :]
        k_copy_rows<<<(Bb * 512 + 255) / 256, 256, 0, stream>>>(
            w.YL, 512, dp + (size_t)(Sl - 1) * 512, (long)Sl * 512, Bb, 512);
        // o = y_last @ out_W^T + out_b : (32, 512) x (64, 512)^T
        gemm(w.YL, out_W, out_b, w.OO, Bb, 64, 512, stream);
        // outs[:, t, :] = o
        k_copy_rows<<<(Bb * 64 + 255) / 256, 256, 0, stream>>>(
            dout + (size_t)t * 64, (long)Tt * 64, w.OO, 64, Bb, 64);
        if (t < Tt - 1) {
            // next emb = o @ emb_W^T + emb_b
            gemm(w.OO, emb_W, emb_b, w.TE, Bb, 512, 64, stream);
            k_copy_rows<<<(Bb * 512 + 255) / 256, 256, 0, stream>>>(
                w.CUR + (size_t)(t + 1) * 512, (long)Tt * 512, w.TE, 512, Bb, 512);
        }
    }
}
